// Transformer_5454608466766
// MI455X (gfx1250) — matrix operand order repaired and checked by reading, not compiled, not run on hardware
//
#include <hip/hip_runtime.h>
#include <hip/hip_bf16.h>
#include <math.h>

typedef __attribute__((ext_vector_type(16))) _Float16 v16h;
typedef __attribute__((ext_vector_type(8)))  _Float16 v8h;
typedef __attribute__((ext_vector_type(8)))  float    v8f;

constexpr int B_ = 4, S_ = 1024, DM_ = 1024, H_ = 16, HD_ = 64, I_ = 4096;
constexpr int MAXREL_ = 10, NREL_ = 2 * MAXREL_ + 1;
constexpr float LN_EPS_ = 1e-3f;
constexpr int ROWS_ = B_ * S_;    // 4096 token rows
constexpr int LDP = 40;           // LDS pitch in halfs (32 + 8 pad, keeps 16B align)

// ---------------- workspace layout (bytes) ----------------
constexpr size_t SZ_XH     = (size_t)ROWS_ * DM_ * 2;     // f16 activation [4096,1024]
constexpr size_t SZ_W1     = (size_t)DM_ * DM_ * 2;       // f16 1024x1024 weight
constexpr size_t SZ_WI     = (size_t)DM_ * I_ * 2;        // f16 1024x4096 weight
constexpr size_t SZ_RELD   = (size_t)B_ * H_ * S_ * NREL_ * 4;
constexpr size_t SZ_F32ACT = (size_t)ROWS_ * DM_ * 4;

constexpr size_t OFF_XH    = 0;
constexpr size_t OFF_WQK   = OFF_XH    + SZ_XH;
constexpr size_t OFF_WV    = OFF_WQK   + SZ_W1;
constexpr size_t OFF_WAP   = OFF_WV    + SZ_W1;
constexpr size_t OFF_WI    = OFF_WAP   + SZ_W1;
constexpr size_t OFF_WO    = OFF_WI    + SZ_WI;
constexpr size_t OFF_QKH   = OFF_WO    + SZ_WI;
constexpr size_t OFF_VH    = OFF_QKH   + SZ_XH;
constexpr size_t OFF_RELD  = OFF_VH    + SZ_XH;
constexpr size_t OFF_CTXH  = OFF_RELD  + SZ_RELD;
constexpr size_t OFF_PRE   = OFF_CTXH  + SZ_XH;     // reused for both pre-LN buffers
constexpr size_t OFF_AOUTF = OFF_PRE   + SZ_F32ACT;
constexpr size_t OFF_AOUTH = OFF_AOUTF + SZ_F32ACT;
constexpr size_t OFF_INTER = OFF_AOUTH + SZ_XH;     // f16 [4096,4096]

// ---------------- small device helpers ----------------
__device__ inline float gelu_f(float x) {
    float x3 = x * x * x;
    return 0.5f * x * (1.0f + tanhf(0.7978845608028654f * (x + 0.044715f * x3)));
}

__device__ inline float wave_max(float v) {
#pragma unroll
    for (int o = 16; o > 0; o >>= 1) v = fmaxf(v, __shfl_xor(v, o, 32));
    return v;
}
__device__ inline float wave_sum(float v) {
#pragma unroll
    for (int o = 16; o > 0; o >>= 1) v += __shfl_xor(v, o, 32);
    return v;
}

// A fragment: 16x32 f16 tile, ISA 7.12.2 layout.
// lane 0-15 = row M, K = {kh..kh+7, 16+kh..16+kh+7}, kh = (lane>>4)*8
__device__ inline v16h load_frag_a(const _Float16* ls, int lane, int rowbase) {
    int row = rowbase + (lane & 15);
    int kh  = (lane >> 4) << 3;
    const _Float16* p = ls + row * LDP + kh;
    union { v16h v; v8h h[2]; } u;
    u.h[0] = *(const v8h*)(p);
    u.h[1] = *(const v8h*)(p + 16);
    return u.v;
}
// B fragment from Bt stored [N][K] in LDS: lane&15 = col N, K = {kh..kh+7, 16+kh..16+kh+7}, kh = (lane>>4)*8
__device__ inline v16h load_frag_b(const _Float16* ls, int lane, int nb) {
    int col = nb + (lane & 15);
    int kh  = (lane >> 4) << 3;
    const _Float16* p = ls + col * LDP + kh;
    union { v16h v; v8h h[2]; } u;
    u.h[0] = *(const v8h*)(p);
    u.h[1] = *(const v8h*)(p + 16);
    return u.v;
}

// ---------------- conversion kernels ----------------
__global__ void cvt_f32_to_f16(const float* __restrict__ src, _Float16* __restrict__ dst, size_t n) {
    size_t i = (size_t)blockIdx.x * blockDim.x + threadIdx.x;
    if (i < n) dst[i] = (_Float16)src[i];
}

// dst[n*K + k] = src[k*N + n]  (weight W[K][N] fp32 -> Wt[N][K] f16)
__global__ void transpose_to_f16(const float* __restrict__ src, _Float16* __restrict__ dst,
                                 int K, int N) {
    size_t i = (size_t)blockIdx.x * blockDim.x + threadIdx.x;
    if (i >= (size_t)K * N) return;
    int k = (int)(i % K);
    int n = (int)(i / K);
    dst[i] = (_Float16)src[(size_t)k * N + n];
}

// ---------------- generic WMMA GEMM: C[M,N] = A[M,K](f16) * Bt[N,K](f16) ----------------
// Block tile 128x64, 256 threads (8 wave32), ping-pong LDS, register prefetch.
// EPI 0: outH = C                (projections, with bias)
// EPI 1: outF = gelu(C) + resid  (attn proj / FFN2 pre-LN)
// EPI 2: outH = gelu(C)          (FFN1 -> inter, f16)
template <int EPI>
__global__ __launch_bounds__(256) void gemm_wmma(
        const _Float16* __restrict__ A, int lda,
        const _Float16* __restrict__ Bt, int ldb,
        const float* __restrict__ bias,
        const float* __restrict__ resid,
        float* __restrict__ outF, _Float16* __restrict__ outH,
        int ldc, int K) {
    __shared__ _Float16 lsA[2][128 * LDP];
    __shared__ _Float16 lsB[2][64 * LDP];
    int t = threadIdx.x, lane = t & 31, wave = t >> 5;
    int nb = (wave & 3) * 16;      // n-strip within 64
    int mh = (wave >> 2) * 64;     // m-half within 128
    int m0 = blockIdx.y * 128, n0 = blockIdx.x * 64;

    // cooperative loader geometry (chunks of 8 halfs = 16B)
    int ra0 = t >> 2,         ca0 = (t & 3) << 3;           // A rows 0..63
    int ra1 = (t + 256) >> 2, ca1 = ((t + 256) & 3) << 3;   // A rows 64..127
    int rb  = t >> 2,         cb  = (t & 3) << 3;           // B rows 0..63
    const _Float16* Ar0 = A  + (size_t)(m0 + ra0) * lda + ca0;
    const _Float16* Ar1 = A  + (size_t)(m0 + ra1) * lda + ca1;
    const _Float16* Br  = Bt + (size_t)(n0 + rb)  * ldb + cb;

    v8h pa0 = *(const v8h*)(Ar0);
    v8h pa1 = *(const v8h*)(Ar1);
    v8h pb  = *(const v8h*)(Br);

    int gn = n0 + nb + (lane & 15);
    float bv = bias ? bias[gn] : 0.0f;
    v8f acc[4];
#pragma unroll
    for (int mt = 0; mt < 4; ++mt) acc[mt] = (v8f){bv, bv, bv, bv, bv, bv, bv, bv};

    for (int k0 = 0; k0 < K; k0 += 32) {
        int buf = (k0 >> 5) & 1;
        *(v8h*)(&lsA[buf][ra0 * LDP + ca0]) = pa0;
        *(v8h*)(&lsA[buf][ra1 * LDP + ca1]) = pa1;
        *(v8h*)(&lsB[buf][rb  * LDP + cb])  = pb;
        __syncthreads();
        int k1 = k0 + 32;
        if (k1 < K) {                       // register prefetch of next chunk
            pa0 = *(const v8h*)(Ar0 + k1);
            pa1 = *(const v8h*)(Ar1 + k1);
            pb  = *(const v8h*)(Br  + k1);
            if ((t & 3) == 0 && k1 + 32 < K) {   // global_prefetch_b8, 2 chunks ahead
                __builtin_prefetch(Ar0 + k1 + 32, 0, 1);
                __builtin_prefetch(Ar1 + k1 + 32, 0, 1);
                __builtin_prefetch(Br  + k1 + 32, 0, 1);
            }
        }
        v16h bf = load_frag_b(lsB[buf], lane, nb);
#pragma unroll
        for (int mt = 0; mt < 4; ++mt) {
            v16h af = load_frag_a(lsA[buf], lane, mh + mt * 16);
            acc[mt] = __builtin_amdgcn_wmma_f32_16x16x32_f16(
                false, af, false, bf, (short)0, acc[mt], false, false);
        }
    }

#pragma unroll
    for (int mt = 0; mt < 4; ++mt) {
#pragma unroll
        for (int r = 0; r < 8; ++r) {
            int gm = m0 + mh + mt * 16 + ((lane >> 4) << 3) + r;
            float c = acc[mt][r];
            size_t o = (size_t)gm * ldc + gn;
            if (EPI == 0) {
                outH[o] = (_Float16)c;
            } else if (EPI == 1) {
                outF[o] = gelu_f(c) + resid[o];
            } else {
                outH[o] = (_Float16)gelu_f(c);
            }
        }
    }
}

// ---------------- reldot: reldot[z][q][r] = qk[b,q,h*64+:] . rel_emb[r,:] ----------------
__global__ void reldot_kernel(const _Float16* __restrict__ qk,
                              const float* __restrict__ rel_emb,
                              float* __restrict__ reldot) {
    size_t i = (size_t)blockIdx.x * blockDim.x + threadIdx.x;
    if (i >= (size_t)B_ * H_ * S_ * NREL_) return;
    int r = (int)(i % NREL_);
    size_t zq = i / NREL_;
    int q = (int)(zq % S_);
    int z = (int)(zq / S_);
    int b = z >> 4, h = z & 15;
    const _Float16* qp = qk + ((size_t)b * S_ + q) * DM_ + h * HD_;
    const float* rp = rel_emb + (size_t)r * HD_;
    float s = 0.0f;
#pragma unroll 8
    for (int d = 0; d < HD_; ++d) s += (float)qp[d] * rp[d];
    reldot[i] = s;
}

// ---------------- scores: per (b,h) Q*K^T + rel + mask -> fp32 probs buffer ----------------
// Block tile 128(q) x 64(k), 256 threads, K=HD=64 (2 pipelined chunks)
__global__ __launch_bounds__(256) void attn_scores_kernel(
        const _Float16* __restrict__ qk,
        const float* __restrict__ reldot,
        const int* __restrict__ mask,
        float* __restrict__ probs) {
    __shared__ _Float16 lsA[2][128 * LDP];
    __shared__ _Float16 lsB[2][64 * LDP];
    int t = threadIdx.x, lane = t & 31, wave = t >> 5;
    int nb = (wave & 3) * 16;
    int mh = (wave >> 2) * 64;
    int z = blockIdx.z, b = z >> 4, h = z & 15;
    int m0 = blockIdx.y * 128, n0 = blockIdx.x * 64;
    const _Float16* Ab = qk + (size_t)b * S_ * DM_ + h * HD_;   // rows, stride DM_

    int ra0 = t >> 2,         ca0 = (t & 3) << 3;
    int ra1 = (t + 256) >> 2, ca1 = ((t + 256) & 3) << 3;
    int rb  = t >> 2,         cb  = (t & 3) << 3;
    const _Float16* Ar0 = Ab + (size_t)(m0 + ra0) * DM_ + ca0;
    const _Float16* Ar1 = Ab + (size_t)(m0 + ra1) * DM_ + ca1;
    const _Float16* Br  = Ab + (size_t)(n0 + rb)  * DM_ + cb;

    v8h pa0 = *(const v8h*)(Ar0);
    v8h pa1 = *(const v8h*)(Ar1);
    v8h pb  = *(const v8h*)(Br);

    v8f acc[4];
#pragma unroll
    for (int mt = 0; mt < 4; ++mt) acc[mt] = (v8f){0, 0, 0, 0, 0, 0, 0, 0};

#pragma unroll
    for (int k0 = 0; k0 < HD_; k0 += 32) {
        int buf = (k0 >> 5) & 1;
        *(v8h*)(&lsA[buf][ra0 * LDP + ca0]) = pa0;
        *(v8h*)(&lsA[buf][ra1 * LDP + ca1]) = pa1;
        *(v8h*)(&lsB[buf][rb  * LDP + cb])  = pb;
        __syncthreads();
        int k1 = k0 + 32;
        if (k1 < HD_) {
            pa0 = *(const v8h*)(Ar0 + k1);
            pa1 = *(const v8h*)(Ar1 + k1);
            pb  = *(const v8h*)(Br  + k1);
        }
        v16h bf = load_frag_b(lsB[buf], lane, nb);
#pragma unroll
        for (int mt = 0; mt < 4; ++mt) {
            v16h af = load_frag_a(lsA[buf], lane, mh + mt * 16);
            acc[mt] = __builtin_amdgcn_wmma_f32_16x16x32_f16(
                false, af, false, bf, (short)0, acc[mt], false, false);
        }
    }

    int gn = n0 + nb + (lane & 15);                 // key index k
    float mval = (float)mask[(size_t)b * S_ + gn];  // column mask, constant per lane
    float* outz = probs + (size_t)z * S_ * S_;
    const float* rdz = reldot + (size_t)z * S_ * NREL_;
#pragma unroll
    for (int mt = 0; mt < 4; ++mt) {
#pragma unroll
        for (int r = 0; r < 8; ++r) {
            int gm = m0 + mh + mt * 16 + ((lane >> 4) << 3) + r;  // query index q
            int d = gn - gm;
            d = d < -MAXREL_ ? -MAXREL_ : (d > MAXREL_ ? MAXREL_ : d);
            float c = acc[mt][r] + rdz[(size_t)gm * NREL_ + d + MAXREL_];
            float am = (gm == gn) ? 0.0f : mval;                  // self-exclusion
            c += (1.0f - am) * -10000.0f;
            outz[(size_t)gm * S_ + gn] = c;
        }
    }
}

// ---------------- row softmax in place (rows = B*H*S, width S) ----------------
__global__ void softmax_kernel(float* __restrict__ probs) {
    __shared__ float red[4];
    int t = threadIdx.x, lane = t & 31, wave = t >> 5;
    float* p = probs + (size_t)blockIdx.x * S_;

    float lmax = -3.0e38f;
    for (int i = t; i < S_; i += 128) lmax = fmaxf(lmax, p[i]);
    lmax = wave_max(lmax);
    if (lane == 0) red[wave] = lmax;
    __syncthreads();
    float vmax = fmaxf(fmaxf(red[0], red[1]), fmaxf(red[2], red[3]));
    __syncthreads();

    float lsum = 0.0f;
    for (int i = t; i < S_; i += 128) {
        float e = expf(p[i] - vmax);
        p[i] = e;
        lsum += e;
    }
    lsum = wave_sum(lsum);
    if (lane == 0) red[wave] = lsum;
    __syncthreads();
    float inv = 1.0f / (red[0] + red[1] + red[2] + red[3]);
    for (int i = t; i < S_; i += 128) p[i] *= inv;
}

// ---------------- ctx: per (b,h) probs(fp32->f16) @ V -> ctx_h[b,s,h*64+d] ----------------
// Block tile 128(q) x 64(d), 256 threads, pipelined over K=S
__global__ __launch_bounds__(256) void attn_ctx_kernel(
        const float* __restrict__ probs,
        const _Float16* __restrict__ vh,
        _Float16* __restrict__ ctxh) {
    __shared__ _Float16 lsA[2][128 * LDP];
    __shared__ _Float16 lsB[2][64 * LDP];
    int t = threadIdx.x, lane = t & 31, wave = t >> 5;
    int nb = (wave & 3) * 16;
    int mh = (wave >> 2) * 64;
    int z = blockIdx.z, b = z >> 4, h = z & 15;
    int m0 = blockIdx.y * 128;                               // query tile
    const float* Ap = probs + (size_t)z * S_ * S_;
    const _Float16* Bp = vh + (size_t)b * S_ * DM_ + h * HD_;  // B[k][d], stride DM_

    int ra0 = t >> 2,         ca0 = (t & 3) << 3;            // A: 128 rows x 32 (fp32)
    int ra1 = (t + 256) >> 2, ca1 = ((t + 256) & 3) << 3;
    int kr  = t >> 3,         ns  = (t & 7) << 3;            // B: 32 k-rows x 64 d
    const float* Ar0 = Ap + (size_t)(m0 + ra0) * S_ + ca0;
    const float* Ar1 = Ap + (size_t)(m0 + ra1) * S_ + ca1;
    const _Float16* Bkr = Bp + (size_t)kr * DM_ + ns;

    float4 fa0 = *(const float4*)(Ar0);
    float4 fa1 = *(const float4*)(Ar0 + 4);
    float4 fb0 = *(const float4*)(Ar1);
    float4 fb1 = *(const float4*)(Ar1 + 4);
    v8h pb = *(const v8h*)(Bkr);

    v8f acc[4];
#pragma unroll
    for (int mt = 0; mt < 4; ++mt) acc[mt] = (v8f){0, 0, 0, 0, 0, 0, 0, 0};

    for (int k0 = 0; k0 < S_; k0 += 32) {
        int buf = (k0 >> 5) & 1;
        {   // commit prefetched A (fp32 -> f16) and transposed B to LDS
            _Float16* dp = &lsA[buf][ra0 * LDP + ca0];
            dp[0] = (_Float16)fa0.x; dp[1] = (_Float16)fa0.y;
            dp[2] = (_Float16)fa0.z; dp[3] = (_Float16)fa0.w;
            dp[4] = (_Float16)fa1.x; dp[5] = (_Float16)fa1.y;
            dp[6] = (_Float16)fa1.z; dp[7] = (_Float16)fa1.w;
            dp = &lsA[buf][ra1 * LDP + ca1];
            dp[0] = (_Float16)fb0.x; dp[1] = (_Float16)fb0.y;
            dp[2] = (_Float16)fb0.z; dp[3] = (_Float16)fb0.w;
            dp[4] = (_Float16)fb1.x; dp[5] = (_Float16)fb1.y;
            dp[6] = (_Float16)fb1.z; dp[7] = (_Float16)fb1.w;
#pragma unroll
            for (int j = 0; j < 8; ++j) lsB[buf][(ns + j) * LDP + kr] = pb[j];
        }
        __syncthreads();
        int k1 = k0 + 32;
        if (k1 < S_) {
            fa0 = *(const float4*)(Ar0 + k1);
            fa1 = *(const float4*)(Ar0 + k1 + 4);
            fb0 = *(const float4*)(Ar1 + k1);
            fb1 = *(const float4*)(Ar1 + k1 + 4);
            pb  = *(const v8h*)(Bkr + (size_t)k1 * DM_);
            if ((t & 3) == 0 && k1 + 32 < S_) {
                __builtin_prefetch(Ar0 + k1 + 32, 0, 1);
                __builtin_prefetch(Ar1 + k1 + 32, 0, 1);
            }
        }
        v16h bf = load_frag_b(lsB[buf], lane, nb);
#pragma unroll
        for (int mt = 0; mt < 4; ++mt) {
            v16h af = load_frag_a(lsA[buf], lane, mh + mt * 16);
            acc[mt] = __builtin_amdgcn_wmma_f32_16x16x32_f16(
                false, af, false, bf, (short)0, acc[mt], false, false);
        }
    }

    int gn = nb + (lane & 15);                      // d within head
#pragma unroll
    for (int mt = 0; mt < 4; ++mt) {
#pragma unroll
        for (int r = 0; r < 8; ++r) {
            int gm = m0 + mh + mt * 16 + ((lane >> 4) << 3) + r;  // query index
            ctxh[((size_t)b * S_ + gm) * DM_ + h * HD_ + gn] = (_Float16)acc[mt][r];
        }
    }
}

// ---------------- row layernorm (width DM_), optional f16 copy ----------------
__global__ void layernorm_kernel(const float* __restrict__ in,
                                 const float* __restrict__ g, const float* __restrict__ bta,
                                 float* __restrict__ outF, _Float16* __restrict__ outH) {
    __shared__ float red[8];
    int t = threadIdx.x, lane = t & 31, wave = t >> 5;
    const float* p = in + (size_t)blockIdx.x * DM_;

    float s = 0.0f, s2 = 0.0f;
    for (int i = t; i < DM_; i += 128) {
        float v = p[i];
        s += v; s2 += v * v;
    }
    s = wave_sum(s); s2 = wave_sum(s2);
    if (lane == 0) { red[wave] = s; red[4 + wave] = s2; }
    __syncthreads();
    float mean = (red[0] + red[1] + red[2] + red[3]) * (1.0f / DM_);
    float msq  = (red[4] + red[5] + red[6] + red[7]) * (1.0f / DM_);
    float var  = msq - mean * mean;
    float rstd = rsqrtf(var + LN_EPS_);

    float* oF = outF + (size_t)blockIdx.x * DM_;
    _Float16* oH = outH ? outH + (size_t)blockIdx.x * DM_ : nullptr;
    for (int i = t; i < DM_; i += 128) {
        float y = (p[i] - mean) * rstd * g[i] + bta[i];
        oF[i] = y;
        if (oH) oH[i] = (_Float16)y;
    }
}

// ---------------- launcher ----------------
extern "C" void kernel_launch(void* const* d_in, const int* in_sizes, int n_in,
                              void* d_out, int out_size, void* d_ws, size_t ws_size,
                              hipStream_t stream) {
    const float* x    = (const float*)d_in[0];
    const int*   mask = (const int*)  d_in[1];
    const float* Wqk  = (const float*)d_in[2];
    const float* bqk  = (const float*)d_in[3];
    const float* Wv   = (const float*)d_in[4];
    const float* bv   = (const float*)d_in[5];
    const float* rel  = (const float*)d_in[6];
    const float* Wap  = (const float*)d_in[7];
    const float* bap  = (const float*)d_in[8];
    const float* g1   = (const float*)d_in[9];
    const float* b1   = (const float*)d_in[10];
    const float* Wi   = (const float*)d_in[11];
    const float* bi   = (const float*)d_in[12];
    const float* Wo   = (const float*)d_in[13];
    const float* bo   = (const float*)d_in[14];
    const float* g2   = (const float*)d_in[15];
    const float* b2   = (const float*)d_in[16];

    char* ws = (char*)d_ws;
    _Float16* xh    = (_Float16*)(ws + OFF_XH);
    _Float16* wqkT  = (_Float16*)(ws + OFF_WQK);
    _Float16* wvT   = (_Float16*)(ws + OFF_WV);
    _Float16* wapT  = (_Float16*)(ws + OFF_WAP);
    _Float16* wiT   = (_Float16*)(ws + OFF_WI);
    _Float16* woT   = (_Float16*)(ws + OFF_WO);
    _Float16* qkh   = (_Float16*)(ws + OFF_QKH);
    _Float16* vh    = (_Float16*)(ws + OFF_VH);
    float*    reld  = (float*)   (ws + OFF_RELD);
    _Float16* ctxh  = (_Float16*)(ws + OFF_CTXH);
    float*    pre   = (float*)   (ws + OFF_PRE);
    float*    aoutF = (float*)   (ws + OFF_AOUTF);
    _Float16* aoutH = (_Float16*)(ws + OFF_AOUTH);
    _Float16* interH= (_Float16*)(ws + OFF_INTER);

    float* outp  = (float*)d_out;                         // [B,S,DM]
    float* probs = outp + (size_t)ROWS_ * DM_;            // [B,H,S,S]

    dim3 blk128(128), blk256(256);

    // --- precision conversions ---
    {
        size_t n = (size_t)ROWS_ * DM_;
        cvt_f32_to_f16<<<dim3((unsigned)((n + 255) / 256)), 256, 0, stream>>>(x, xh, n);
    }
    {
        size_t n = (size_t)DM_ * DM_;
        unsigned g = (unsigned)((n + 255) / 256);
        transpose_to_f16<<<g, 256, 0, stream>>>(Wqk, wqkT, DM_, DM_);
        transpose_to_f16<<<g, 256, 0, stream>>>(Wv,  wvT,  DM_, DM_);
        transpose_to_f16<<<g, 256, 0, stream>>>(Wap, wapT, DM_, DM_);
        size_t nw = (size_t)DM_ * I_;
        unsigned gw = (unsigned)((nw + 255) / 256);
        transpose_to_f16<<<gw, 256, 0, stream>>>(Wi, wiT, DM_, I_);   // -> [I][DM]
        transpose_to_f16<<<gw, 256, 0, stream>>>(Wo, woT, I_, DM_);   // -> [DM][I]
    }

    // --- QK and V projections ---
    gemm_wmma<0><<<dim3(DM_ / 64, ROWS_ / 128), blk256, 0, stream>>>(
        xh, DM_, wqkT, DM_, bqk, nullptr, nullptr, qkh, DM_, DM_);
    gemm_wmma<0><<<dim3(DM_ / 64, ROWS_ / 128), blk256, 0, stream>>>(
        xh, DM_, wvT, DM_, bv, nullptr, nullptr, vh, DM_, DM_);

    // --- relative-position dot table ---
    {
        size_t n = (size_t)B_ * H_ * S_ * NREL_;
        reldot_kernel<<<dim3((unsigned)((n + 255) / 256)), 256, 0, stream>>>(qkh, rel, reld);
    }

    // --- attention ---
    attn_scores_kernel<<<dim3(S_ / 64, S_ / 128, B_ * H_), blk256, 0, stream>>>(
        qkh, reld, mask, probs);
    softmax_kernel<<<dim3(B_ * H_ * S_), blk128, 0, stream>>>(probs);
    attn_ctx_kernel<<<dim3(1, S_ / 128, B_ * H_), blk256, 0, stream>>>(probs, vh, ctxh);

    // --- attention projection + residual + LN1 ---
    gemm_wmma<1><<<dim3(DM_ / 64, ROWS_ / 128), blk256, 0, stream>>>(
        ctxh, DM_, wapT, DM_, bap, x, pre, nullptr, DM_, DM_);
    layernorm_kernel<<<dim3(ROWS_), blk128, 0, stream>>>(pre, g1, b1, aoutF, aoutH);

    // --- FFN ---
    gemm_wmma<2><<<dim3(I_ / 64, ROWS_ / 128), blk256, 0, stream>>>(
        aoutH, DM_, wiT, DM_, bi, nullptr, nullptr, interH, I_, DM_);
    gemm_wmma<1><<<dim3(DM_ / 64, ROWS_ / 128), blk256, 0, stream>>>(
        interH, I_, woT, I_, bo, aoutF, pre, nullptr, DM_, I_);
    layernorm_kernel<<<dim3(ROWS_), blk128, 0, stream>>>(pre, g2, b2, outp, nullptr);
}
